// Attention_6116033429789
// MI455X (gfx1250) — compile-verified, matrix operand order checked by reading, not run on hardware
//
#include <hip/hip_runtime.h>

#define H 768
#define NHEADS 12
#define HDIM 64

typedef __attribute__((ext_vector_type(16))) __bf16 v16bf;
typedef __attribute__((ext_vector_type(8)))  __bf16 v8bf;
typedef __attribute__((ext_vector_type(8)))  float  v8f;

union BF16Frag { v16bf v; v8bf h[2]; };
union BF16x4   { uint2 u2; __bf16 b[4]; };

// Per CDNA5 ISA 7.12.2 (16-bit A 16x32): lane&15 = row, lane>>4 selects the
// K sub-group; element e -> k = (e>>3)*16 + (lane>>4)*8 + (e&7). Both 8-element
// runs are contiguous, so two b128 loads build a fragment.
static __device__ __forceinline__ v16bf ld_frag(const __bf16* lo, const __bf16* hi) {
  BF16Frag u;
  u.h[0] = *reinterpret_cast<const v8bf*>(lo);
  u.h[1] = *reinterpret_cast<const v8bf*>(hi);
  return u.v;
}

static __device__ __forceinline__ v8f wmma_bf16(v16bf a, v16bf b, v8f c) {
  return __builtin_amdgcn_wmma_f32_16x16x32_bf16(false, a, false, b, (short)0, c, false, false);
}

// Async copy of 16 contiguous bytes global -> LDS (per lane), ASYNCcnt-tracked.
static __device__ __forceinline__ void async_g2l_b128(unsigned ldsOff, const __bf16* g) {
  asm volatile("global_load_async_to_lds_b128 %0, %1, off"
               :: "v"(ldsOff), "v"(g) : "memory");
}
static __device__ __forceinline__ void wait_async0() {
  asm volatile("s_wait_asynccnt 0x0" ::: "memory");
}

// ---------------------------------------------------------------------------
// GEMM: Out[M x 768] = A[M x 768] (fp32) @ W[768 x 768] (fp32) + bias
// BF16OUT selects bf16 (workspace) vs fp32 (d_out). grid=(768/64, M/64), block 128.
// ---------------------------------------------------------------------------
template <bool BF16OUT>
__global__ __launch_bounds__(128)
void proj_gemm_kernel(const float* __restrict__ A, const float* __restrict__ W,
                      const float* __restrict__ bias, void* __restrict__ outP) {
  __shared__ __bf16 aLds[64 * 48];   // [row][k'], stride 48 (96B rows, 16B aligned)
  __shared__ __bf16 wLds[64 * 48];   // [col][k'] (W chunk transposed)

  const int t    = threadIdx.x;
  const int wave = t >> 5;
  const int lane = t & 31;
  const int m16  = lane & 15;
  const int hi   = lane >> 4;
  const int tileM = blockIdx.y * 64;
  const int tileN = blockIdx.x * 64;

  v8f c[4] = { {0,0,0,0,0,0,0,0}, {0,0,0,0,0,0,0,0},
               {0,0,0,0,0,0,0,0}, {0,0,0,0,0,0,0,0} };

  for (int kk = 0; kk < H; kk += 32) {
    __syncthreads();
    // A chunk 64x32: float4 loads, packed b64 LDS stores
    #pragma unroll
    for (int i = 0; i < 4; ++i) {
      int idx = t + 128 * i;                 // 0..511, 4 elements each
      int r = idx >> 3, kp4 = (idx & 7) * 4;
      float4 v = *reinterpret_cast<const float4*>(&A[(size_t)(tileM + r) * H + kk + kp4]);
      BF16x4 p;
      p.b[0] = (__bf16)v.x; p.b[1] = (__bf16)v.y;
      p.b[2] = (__bf16)v.z; p.b[3] = (__bf16)v.w;
      *reinterpret_cast<uint2*>(&aLds[r * 48 + kp4]) = p.u2;
    }
    // W chunk 32x64: float4 loads along n, transposed scalar stores
    #pragma unroll
    for (int i = 0; i < 4; ++i) {
      int idx = t + 128 * i;                 // 0..511
      int kp = idx >> 4, n4 = (idx & 15) * 4;
      float4 v = *reinterpret_cast<const float4*>(&W[(size_t)(kk + kp) * H + tileN + n4]);
      wLds[(n4 + 0) * 48 + kp] = (__bf16)v.x;
      wLds[(n4 + 1) * 48 + kp] = (__bf16)v.y;
      wLds[(n4 + 2) * 48 + kp] = (__bf16)v.z;
      wLds[(n4 + 3) * 48 + kp] = (__bf16)v.w;
    }
    __syncthreads();

    const __bf16* aRow = &aLds[(wave * 16 + m16) * 48 + hi * 8];
    v16bf aF = ld_frag(aRow, aRow + 16);
    #pragma unroll
    for (int nt = 0; nt < 4; ++nt) {
      const __bf16* bRow = &wLds[(nt * 16 + m16) * 48 + hi * 8];
      v16bf bF = ld_frag(bRow, bRow + 16);
      c[nt] = wmma_bf16(aF, bF, c[nt]);
    }
  }

  #pragma unroll
  for (int nt = 0; nt < 4; ++nt) {
    int col = tileN + nt * 16 + m16;
    float bv = bias[col];
    #pragma unroll
    for (int r = 0; r < 8; ++r) {            // C layout: VGPR r -> row r (+8 hi lanes)
      size_t off = (size_t)(tileM + wave * 16 + r + hi * 8) * H + col;
      float v = c[nt][r] + bv;
      if (BF16OUT) reinterpret_cast<__bf16*>(outP)[off] = (__bf16)v;
      else         reinterpret_cast<float*>(outP)[off]  = v;
    }
  }
}

// ---------------------------------------------------------------------------
// Attention: per (b, h, 32-query block). K tiles async-staged to LDS; scores ->
// softmax (-> probs out for ii) -> P@V. ctx written as 0.5*ctx (+= if accumulate).
// grid = (Nq/32, 12, 8), block 64 (2 waves), dyn LDS = 32*Nk*6 bytes.
// ---------------------------------------------------------------------------
__global__ __launch_bounds__(64)
void attn_kernel(const __bf16* __restrict__ Q, const __bf16* __restrict__ K,
                 const __bf16* __restrict__ V, float* __restrict__ ctx,
                 float* __restrict__ probOut, int Nq, int Nk, int accumulate) {
  extern __shared__ char smemRaw[];
  float*  scoresF = reinterpret_cast<float*>(smemRaw);                        // [32][Nk]
  __bf16* probsB  = reinterpret_cast<__bf16*>(smemRaw + (size_t)32 * Nk * 4); // [32][Nk]
  __shared__ __bf16 kLds[64 * 64];    // 64 keys x 64 dims, row-major (async-staged)
  __shared__ __bf16 vt[64 * 48];      // V chunk transposed: [d][key'], stride 48

  const int t    = threadIdx.x;
  const int wave = t >> 5;
  const int lane = t & 31;
  const int m16  = lane & 15;
  const int hi   = lane >> 4;
  const int h = blockIdx.y;
  const int b = blockIdx.z;
  const int qBase = blockIdx.x * 32;
  const int qRow  = qBase + wave * 16;

  const __bf16* Qb = Q + (size_t)b * Nq * H + (size_t)h * HDIM;
  const __bf16* Kb = K + (size_t)b * Nk * H + (size_t)h * HDIM;
  const __bf16* Vb = V + (size_t)b * Nk * H + (size_t)h * HDIM;

  const unsigned kLdsBase = (unsigned)(uintptr_t)&kLds[0];  // LDS byte offset

  // Q fragments: K-dim 0..31 and 32..63
  const __bf16* qrow = Qb + (size_t)(qRow + m16) * H;
  v16bf aQ0 = ld_frag(qrow + hi * 8,      qrow + 16 + hi * 8);
  v16bf aQ1 = ld_frag(qrow + 32 + hi * 8, qrow + 48 + hi * 8);

  // ---- scores = (Q K^T) * 1/sqrt(64), 64 keys per async-staged LDS tile ----
  for (int kt0 = 0; kt0 < Nk; kt0 += 64) {
    __syncthreads();
    #pragma unroll
    for (int j = 0; j < 8; ++j) {            // 64 rows x 128B = 8KB, 16B per lane
      int idx = t + 64 * j;                  // 0..511
      int row = idx >> 3, chunk = (idx & 7) * 8;
      async_g2l_b128(kLdsBase + (unsigned)(row * 64 + chunk) * 2,
                     Kb + (size_t)(kt0 + row) * H + chunk);
    }
    wait_async0();
    __syncthreads();
    #pragma unroll
    for (int kt2 = 0; kt2 < 4; ++kt2) {
      const __bf16* krow = &kLds[(kt2 * 16 + m16) * 64];
      v16bf bK0 = ld_frag(krow + hi * 8,      krow + 16 + hi * 8);
      v16bf bK1 = ld_frag(krow + 32 + hi * 8, krow + 48 + hi * 8);
      v8f s = {0,0,0,0,0,0,0,0};
      s = wmma_bf16(aQ0, bK0, s);
      s = wmma_bf16(aQ1, bK1, s);
      #pragma unroll
      for (int r = 0; r < 8; ++r)
        scoresF[(size_t)(wave * 16 + r + hi * 8) * Nk + kt0 + kt2 * 16 + m16] = s[r] * 0.125f;
    }
  }
  __syncthreads();

  // ---- softmax: lanes (l, l+16) split row l&15 in half ----
  {
    const int row = wave * 16 + m16;
    float* srow = scoresF + (size_t)row * Nk;
    const int half = Nk >> 1;
    const int c0 = hi * half, c1 = c0 + half;
    float m = -3.0e38f;
    for (int cc = c0; cc < c1; ++cc) m = fmaxf(m, srow[cc]);
    m = fmaxf(m, __shfl_xor(m, 16, 32));
    float sum = 0.f;
    for (int cc = c0; cc < c1; ++cc) sum += __expf(srow[cc] - m);
    sum += __shfl_xor(sum, 16, 32);
    const float inv = 1.0f / sum;
    __bf16* prow = probsB + (size_t)row * Nk;
    if (probOut) {
      float* gp = probOut + (((size_t)b * NHEADS + h) * Nq + qBase + row) * Nk;
      for (int cc = c0; cc < c1; ++cc) {
        float p = __expf(srow[cc] - m) * inv;
        prow[cc] = (__bf16)p;
        gp[cc] = p;
      }
    } else {
      for (int cc = c0; cc < c1; ++cc)
        prow[cc] = (__bf16)(__expf(srow[cc] - m) * inv);
    }
  }
  __syncthreads();

  // ---- ctx = P @ V ----
  v8f c4[4] = { {0,0,0,0,0,0,0,0}, {0,0,0,0,0,0,0,0},
                {0,0,0,0,0,0,0,0}, {0,0,0,0,0,0,0,0} };
  for (int kk = 0; kk < Nk; kk += 32) {
    __syncthreads();
    #pragma unroll
    for (int i = 0; i < 8; ++i) {            // 32x64 V chunk: 8B loads, transpose stores
      int idx = t + 64 * i;                  // 0..511, 4 elements each
      int key = idx >> 4, d0 = (idx & 15) * 4;
      BF16x4 p;
      p.u2 = *reinterpret_cast<const uint2*>(&Vb[(size_t)(kk + key) * H + d0]);
      vt[(d0 + 0) * 48 + key] = p.b[0];
      vt[(d0 + 1) * 48 + key] = p.b[1];
      vt[(d0 + 2) * 48 + key] = p.b[2];
      vt[(d0 + 3) * 48 + key] = p.b[3];
    }
    __syncthreads();
    const __bf16* prow = probsB + (size_t)(wave * 16 + m16) * Nk + kk + hi * 8;
    v16bf aP = ld_frag(prow, prow + 16);
    #pragma unroll
    for (int nt = 0; nt < 4; ++nt) {
      const __bf16* vrow = &vt[(nt * 16 + m16) * 48 + hi * 8];
      v16bf bV = ld_frag(vrow, vrow + 16);
      c4[nt] = wmma_bf16(aP, bV, c4[nt]);
    }
  }

  // ---- 0.5 * ctx, optional accumulate (uniform branch hoisted) ----
  float* cb = ctx + (size_t)b * Nq * H + (size_t)h * HDIM;
  if (accumulate) {
    #pragma unroll
    for (int nt = 0; nt < 4; ++nt) {
      int d = nt * 16 + m16;
      #pragma unroll
      for (int r = 0; r < 8; ++r) {
        size_t off = (size_t)(qRow + r + hi * 8) * H + d;
        cb[off] += c4[nt][r] * 0.5f;
      }
    }
  } else {
    #pragma unroll
    for (int nt = 0; nt < 4; ++nt) {
      int d = nt * 16 + m16;
      #pragma unroll
      for (int r = 0; r < 8; ++r) {
        size_t off = (size_t)(qRow + r + hi * 8) * H + d;
        cb[off] = c4[nt][r] * 0.5f;
      }
    }
  }
}

// ---------------------------------------------------------------------------
extern "C" void kernel_launch(void* const* d_in, const int* in_sizes, int n_in,
                              void* d_out, int out_size, void* d_ws, size_t ws_size,
                              hipStream_t stream) {
  (void)in_sizes; (void)n_in; (void)out_size; (void)ws_size;
  const float* hidden = (const float*)d_in[0];
  const float* text   = (const float*)d_in[1];
  const float* Wq  = (const float*)d_in[2];  const float* bq  = (const float*)d_in[3];
  const float* Wk  = (const float*)d_in[4];  const float* bk  = (const float*)d_in[5];
  const float* Wv  = (const float*)d_in[6];  const float* bv  = (const float*)d_in[7];
  const float* Wqt = (const float*)d_in[8];  const float* bqt = (const float*)d_in[9];
  const float* Wkt = (const float*)d_in[10]; const float* bkt = (const float*)d_in[11];
  const float* Wvt = (const float*)d_in[12]; const float* bvt = (const float*)d_in[13];
  const float* Wo  = (const float*)d_in[14]; const float* bo  = (const float*)d_in[15];

  const int B = 8, Ni = 1024, Nt = 512;
  const size_t szI = (size_t)B * Ni * H;   // 6,291,456 elements
  const size_t szT = (size_t)B * Nt * H;   // 3,145,728 elements

  char* ws = (char*)d_ws;
  __bf16* qI = (__bf16*)ws; ws += szI * 2;
  __bf16* kI = (__bf16*)ws; ws += szI * 2;
  __bf16* vI = (__bf16*)ws; ws += szI * 2;
  __bf16* qT = (__bf16*)ws; ws += szT * 2;
  __bf16* kT = (__bf16*)ws; ws += szT * 2;
  __bf16* vT = (__bf16*)ws; ws += szT * 2;
  float* ctxI = (float*)ws; ws += szI * 4;
  float* ctxT = (float*)ws; ws += szT * 4;

  float* outImg  = (float*)d_out;
  float* outText = outImg + szI;
  float* weights = outText + szT;

  dim3 pb(128);
  dim3 pgI(H / 64, (B * Ni) / 64);   // 12 x 128
  dim3 pgT(H / 64, (B * Nt) / 64);   // 12 x 64

  // QKV projections (fp32 in, bf16 out)
  proj_gemm_kernel<true><<<pgI, pb, 0, stream>>>(hidden, Wq,  bq,  qI);
  proj_gemm_kernel<true><<<pgI, pb, 0, stream>>>(hidden, Wk,  bk,  kI);
  proj_gemm_kernel<true><<<pgI, pb, 0, stream>>>(hidden, Wv,  bv,  vI);
  proj_gemm_kernel<true><<<pgT, pb, 0, stream>>>(text,   Wqt, bqt, qT);
  proj_gemm_kernel<true><<<pgT, pb, 0, stream>>>(text,   Wkt, bkt, kT);
  proj_gemm_kernel<true><<<pgT, pb, 0, stream>>>(text,   Wvt, bvt, vT);

  // Attentions: ctx accumulates 0.5*(ctx_a + ctx_b)
  auto smem = [](int Nk) { return (size_t)32 * Nk * 6; };  // fp32 scores + bf16 probs
  attn_kernel<<<dim3(Ni / 32, NHEADS, B), 64, smem(1024), stream>>>(
      qI, kI, vI, ctxI, weights, Ni, 1024, 0);             // ii (+ probs out)
  attn_kernel<<<dim3(Ni / 32, NHEADS, B), 64, smem(512), stream>>>(
      qI, kT, vT, ctxI, nullptr, Ni, 512, 1);              // it
  attn_kernel<<<dim3(Nt / 32, NHEADS, B), 64, smem(512), stream>>>(
      qT, kT, vT, ctxT, nullptr, Nt, 512, 0);              // tt
  attn_kernel<<<dim3(Nt / 32, NHEADS, B), 64, smem(1024), stream>>>(
      qT, kI, vI, ctxT, nullptr, Nt, 1024, 1);             // ti

  // Output projections (fp32 ctx in, fp32 out)
  proj_gemm_kernel<false><<<pgI, pb, 0, stream>>>(ctxI, Wo, bo, outImg);
  proj_gemm_kernel<false><<<pgT, pb, 0, stream>>>(ctxT, Wo, bo, outText);
}
